// SpectralConv1d_SMM_25262997635104
// MI455X (gfx1250) — compile-verified, matrix operand order checked by reading, not run on hardware
//
#include <hip/hip_runtime.h>
#include <hip/hip_bf16.h>

// SpectralConv1d: B=32, CI=64, CO=64, N=8192, K(modes)=64
#define B_  32
#define CI_ 64
#define CO_ 64
#define N_  8192
#define K_  64

typedef __attribute__((ext_vector_type(16))) __bf16 v16bf;
typedef __attribute__((ext_vector_type(8)))  __bf16 v8bf;
typedef __attribute__((ext_vector_type(8)))  float  v8f;

// ---- WMMA wrapper: D = A(16x32 bf16) x B(32x16 bf16) + C(16x16 f32) ----
__device__ __forceinline__ v8f wmma_bf16(v16bf a, v16bf b, v8f c) {
  return __builtin_amdgcn_wmma_f32_16x16x32_bf16(
      /*neg_a=*/false, a, /*neg_b=*/false, b,
      /*c_mod=*/(short)0, c, /*reuse_a=*/false, /*reuse_b=*/false);
}

// Fragment from a bf16 row. rowk = &row[k0 + half*8] (half = lane>>4).
// Element j -> k = (j<8 ? half*8+j : 16+half*8+(j-8)) rel. to k0: two
// contiguous 8-element (16 B) chunks at rowk and rowk+16. Pure loads, no VALU.
__device__ __forceinline__ v16bf frag_row_bf16(const __bf16* __restrict__ rowk) {
  const v8bf lo = *(const v8bf*)rowk;
  const v8bf hi = *(const v8bf*)(rowk + 16);
  return __builtin_shufflevector(lo, hi, 0, 1, 2, 3, 4, 5, 6, 7,
                                 8, 9, 10, 11, 12, 13, 14, 15);
}

// Fragment from a contiguous f32 row with on-the-fly convert (cold paths only).
__device__ __forceinline__ v16bf frag_row_f32(const float* __restrict__ rowk,
                                              float sgn) {
  const float4 a = ((const float4*)rowk)[0];
  const float4 b = ((const float4*)rowk)[1];
  const float4 c = ((const float4*)(rowk + 16))[0];
  const float4 d = ((const float4*)(rowk + 16))[1];
  v16bf v;
  v[0]  = (__bf16)(sgn * a.x); v[1]  = (__bf16)(sgn * a.y);
  v[2]  = (__bf16)(sgn * a.z); v[3]  = (__bf16)(sgn * a.w);
  v[4]  = (__bf16)(sgn * b.x); v[5]  = (__bf16)(sgn * b.y);
  v[6]  = (__bf16)(sgn * b.z); v[7]  = (__bf16)(sgn * b.w);
  v[8]  = (__bf16)(sgn * c.x); v[9]  = (__bf16)(sgn * c.y);
  v[10] = (__bf16)(sgn * c.z); v[11] = (__bf16)(sgn * c.w);
  v[12] = (__bf16)(sgn * d.x); v[13] = (__bf16)(sgn * d.y);
  v[14] = (__bf16)(sgn * d.z); v[15] = (__bf16)(sgn * d.w);
  return v;
}

// Gather fragment (stage 2 weights only; tiny, L2-resident).
__device__ __forceinline__ v16bf frag_gather_f32(const float* __restrict__ base,
                                                 int strideK, int half, float sgn) {
  v16bf v;
#pragma unroll
  for (int j = 0; j < 16; ++j) {
    const int k = (j < 8) ? (half * 8 + j) : (16 + half * 8 + (j - 8));
    v[j] = (__bf16)(sgn * base[(size_t)k * strideK]);
  }
  return v;
}

// ============================================================================
// Kernel 0: bulk f32 -> bf16 conversion, 8 elements/thread (16 B stores).
// ============================================================================
__global__ __launch_bounds__(256) void k_cvt_bf16(
    const float* __restrict__ src, __bf16* __restrict__ dst, int n8) {
  const int i = blockIdx.x * blockDim.x + threadIdx.x;
  if (i >= n8) return;
  const float4 a = ((const float4*)src)[2 * i];
  const float4 b = ((const float4*)src)[2 * i + 1];
  v8bf o;
  o[0] = (__bf16)a.x; o[1] = (__bf16)a.y; o[2] = (__bf16)a.z; o[3] = (__bf16)a.w;
  o[4] = (__bf16)b.x; o[5] = (__bf16)b.y; o[6] = (__bf16)b.z; o[7] = (__bf16)b.w;
  ((v8bf*)dst)[i] = o;
}

// ============================================================================
// Kernel 1: forward DFT.  xb[b,ci,n] (bf16) x Vf[k,n] (bf16) -> x_ft re/im,
// stored f32 as [kmode][b][ci] so stage-2 A-rows are contiguous.
// One wave per (b, ci-tile, kmode-tile); reduce over N=8192.
// Inner loop: 6 x 16B loads + 2 WMMA, no conversions.
// ============================================================================
__global__ __launch_bounds__(256) void k_fwd_dft(
    const __bf16* __restrict__ xb,
    const __bf16* __restrict__ vfr, const __bf16* __restrict__ vfi,
    float* __restrict__ xr, float* __restrict__ xi) {
  const int lane = threadIdx.x & 31;
  const int wid  = blockIdx.x * (blockDim.x >> 5) + (threadIdx.x >> 5); // 0..511
  const int b    = wid >> 4;         // 32 batches
  const int rt   = (wid >> 2) & 3;   // ci tile
  const int mt   = wid & 3;          // mode tile
  const int l15  = lane & 15;
  const int half = lane >> 4;
  const int ci0  = rt * 16;
  const int km0  = mt * 16;

  const __bf16* arow  = xb  + (size_t)(b * CI_ + ci0 + l15) * N_ + half * 8;
  const __bf16* brrow = vfr + (size_t)(km0 + l15) * N_ + half * 8;
  const __bf16* birow = vfi + (size_t)(km0 + l15) * N_ + half * 8;

  v8f cre = {};
  v8f cim = {};
#pragma unroll 2
  for (int n0 = 0; n0 < N_; n0 += 32) {
    if (n0 + 1024 < N_) __builtin_prefetch(arow + n0 + 1024, 0, 0);
    const v16bf A  = frag_row_bf16(arow  + n0);
    const v16bf Br = frag_row_bf16(brrow + n0);
    const v16bf Bi = frag_row_bf16(birow + n0);
    cre = wmma_bf16(A, Br, cre);
    cim = wmma_bf16(A, Bi, cim);
  }

  // C element (VGPR r, lane) -> ci = ci0 + 8*half + r, kmode = km0 + l15.
  float* dr = xr + (size_t)(km0 + l15) * (B_ * CI_) + b * CI_ + ci0 + half * 8;
  float* di = xi + (size_t)(km0 + l15) * (B_ * CI_) + b * CI_ + ci0 + half * 8;
  ((float4*)dr)[0] = make_float4(cre[0], cre[1], cre[2], cre[3]);
  ((float4*)dr)[1] = make_float4(cre[4], cre[5], cre[6], cre[7]);
  ((float4*)di)[0] = make_float4(cim[0], cim[1], cim[2], cim[3]);
  ((float4*)di)[1] = make_float4(cim[4], cim[5], cim[6], cim[7]);
}

// ============================================================================
// Kernel 2: per-mode complex channel mixing (tiny: ~67 MFLOP total).
// A = x_ft[:, :, k] as [b, ci] (contiguous ci), B = w[:, :, k] gathered.
// out_ft stored f32 [b][o][k] so stage-3 A-rows are contiguous in k.
// ============================================================================
__global__ __launch_bounds__(256) void k_modemix(
    const float* __restrict__ xr, const float* __restrict__ xi,
    const float* __restrict__ wr, const float* __restrict__ wi,
    float* __restrict__ outr, float* __restrict__ outi) {
  const int lane = threadIdx.x & 31;
  const int wid  = blockIdx.x * (blockDim.x >> 5) + (threadIdx.x >> 5); // 0..511
  const int k    = wid >> 3;        // 64 modes
  const int bt   = (wid >> 2) & 1;  // 2 batch tiles
  const int ot   = wid & 3;         // 4 out-channel tiles
  const int l15  = lane & 15;
  const int half = lane >> 4;
  const int b0   = bt * 16;
  const int o0   = ot * 16;

  const float* arow_r = xr + (size_t)k * (B_ * CI_) + (b0 + l15) * CI_ + half * 8;
  const float* arow_i = xi + (size_t)k * (B_ * CI_) + (b0 + l15) * CI_ + half * 8;
  const float* wrb = wr + (size_t)(o0 + l15) * K_ + k;
  const float* wib = wi + (size_t)(o0 + l15) * K_ + k;

  v8f cre = {};
  v8f cim = {};
#pragma unroll
  for (int i0 = 0; i0 < CI_; i0 += 32) {
    const v16bf Ar  = frag_row_f32(arow_r + i0, 1.0f);
    const v16bf Ai  = frag_row_f32(arow_i + i0, 1.0f);
    const v16bf Br  = frag_gather_f32(wrb + (size_t)i0 * (CO_ * K_), CO_ * K_, half,  1.0f);
    const v16bf Bi  = frag_gather_f32(wib + (size_t)i0 * (CO_ * K_), CO_ * K_, half,  1.0f);
    const v16bf Bin = frag_gather_f32(wib + (size_t)i0 * (CO_ * K_), CO_ * K_, half, -1.0f);
    cre = wmma_bf16(Ar, Br,  cre);   // + xr*wr
    cre = wmma_bf16(Ai, Bin, cre);   // - xi*wi
    cim = wmma_bf16(Ar, Bi,  cim);   // + xr*wi
    cim = wmma_bf16(Ai, Br,  cim);   // + xi*wr
  }

  // C element -> b = b0 + 8*half + r, o = o0 + l15; store to [b][o][k].
#pragma unroll
  for (int r = 0; r < 8; ++r) {
    const size_t off = (size_t)(b0 + 8 * half + r) * (CO_ * K_) + (o0 + l15) * K_ + k;
    outr[off] = cre[r];
    outi[off] = cim[r];
  }
}

// ============================================================================
// Kernel 3: inverse DFT, real part only.
// out[b,o,n] = sum_k re[b,o,k]*Vi_r[n,k] - im[b,o,k]*Vi_i[n,k]
// A' = [re | -im] (64x128) preloaded once per wave (f32->bf16, cold),
// B't[n,:] = [Vi_r[n,:] | Vi_i[n,:]] streamed as bf16: 8 x 16B loads +
// 4 WMMA per 16x16 output tile.
// ============================================================================
__global__ __launch_bounds__(256) void k_inv_dft(
    const float* __restrict__ outr, const float* __restrict__ outi,
    const __bf16* __restrict__ vir, const __bf16* __restrict__ vii,
    float* __restrict__ out) {
  const int lane = threadIdx.x & 31;
  const int wid  = blockIdx.x * (blockDim.x >> 5) + (threadIdx.x >> 5); // 0..4095
  const int b    = wid >> 7;          // 32 batches
  const int ot   = (wid >> 5) & 3;    // 4 o tiles
  const int nc   = wid & 31;          // 32 chunks of 256 along n
  const int l15  = lane & 15;
  const int half = lane >> 4;
  const int o0   = ot * 16;
  const int nb   = nc * 256;

  const float* ar = outr + (size_t)(b * CO_ + o0 + l15) * K_ + half * 8;
  const float* ai = outi + (size_t)(b * CO_ + o0 + l15) * K_ + half * 8;
  const v16bf A0 = frag_row_f32(ar,       1.0f);   // re,  k 0..31
  const v16bf A1 = frag_row_f32(ar + 32,  1.0f);   // re,  k 32..63
  const v16bf A2 = frag_row_f32(ai,      -1.0f);   // -im, k 0..31
  const v16bf A3 = frag_row_f32(ai + 32, -1.0f);   // -im, k 32..63

#pragma unroll 4
  for (int t = 0; t < 16; ++t) {
    const int n0 = nb + t * 16;
    const __bf16* vrow_r = vir + (size_t)(n0 + l15) * K_ + half * 8;
    const __bf16* vrow_i = vii + (size_t)(n0 + l15) * K_ + half * 8;
    const v16bf B0 = frag_row_bf16(vrow_r);
    const v16bf B1 = frag_row_bf16(vrow_r + 32);
    const v16bf B2 = frag_row_bf16(vrow_i);
    const v16bf B3 = frag_row_bf16(vrow_i + 32);
    v8f c = {};
    c = wmma_bf16(A0, B0, c);
    c = wmma_bf16(A1, B1, c);
    c = wmma_bf16(A2, B2, c);
    c = wmma_bf16(A3, B3, c);
    // C element -> o = o0 + 8*half + r, n = n0 + l15.
#pragma unroll
    for (int r = 0; r < 8; ++r) {
      out[(size_t)(b * CO_ + o0 + 8 * half + r) * N_ + n0 + l15] = c[r];
    }
  }
}

// ============================================================================
extern "C" void kernel_launch(void* const* d_in, const int* in_sizes, int n_in,
                              void* d_out, int out_size, void* d_ws, size_t ws_size,
                              hipStream_t stream) {
  // setup_inputs() order: x, wr, wi, Vf_r, Vf_i, Vi_r, Vi_i  (all float32)
  const float* x    = (const float*)d_in[0];
  const float* wr   = (const float*)d_in[1];
  const float* wi   = (const float*)d_in[2];
  const float* Vf_r = (const float*)d_in[3];
  const float* Vf_i = (const float*)d_in[4];
  const float* Vi_r = (const float*)d_in[5];
  const float* Vi_i = (const float*)d_in[6];
  float* out = (float*)d_out;

  // Workspace layout (bytes, all 256-aligned):
  //   xb   : bf16 x              [B,CI,N]   33,554,432
  //   vfr/vfi : bf16 Vf          [K,N]       1,048,576 each
  //   vir/vii : bf16 Vi          [N,K]       1,048,576 each
  //   xr/xi   : f32 x_ft         [K,B,CI]      524,288 each
  //   or/oi   : f32 out_ft       [B,CO,K]      524,288 each
  char* p = (char*)d_ws;
  __bf16* xb    = (__bf16*)(p);
  __bf16* vfr_b = (__bf16*)(p + 33554432);
  __bf16* vfi_b = (__bf16*)(p + 34603008);
  __bf16* vir_b = (__bf16*)(p + 35651584);
  __bf16* vii_b = (__bf16*)(p + 36700160);
  float*  xr_ws = (float*)(p + 37748736);
  float*  xi_ws = (float*)(p + 38273024);
  float*  or_ws = (float*)(p + 38797312);
  float*  oi_ws = (float*)(p + 39321600);

  // Stage 0: preconvert streamed operands to bf16 (one pass each).
  const int x_n8 = (B_ * CI_ * N_) / 8;   // 2,097,152
  const int v_n8 = (K_ * N_) / 8;         //    65,536
  k_cvt_bf16<<<x_n8 / 256, 256, 0, stream>>>(x, xb, x_n8);
  k_cvt_bf16<<<v_n8 / 256, 256, 0, stream>>>(Vf_r, vfr_b, v_n8);
  k_cvt_bf16<<<v_n8 / 256, 256, 0, stream>>>(Vf_i, vfi_b, v_n8);
  k_cvt_bf16<<<v_n8 / 256, 256, 0, stream>>>(Vi_r, vir_b, v_n8);
  k_cvt_bf16<<<v_n8 / 256, 256, 0, stream>>>(Vi_i, vii_b, v_n8);

  // Stage 1: 512 waves = 64 blocks x 256 threads
  k_fwd_dft<<<64, 256, 0, stream>>>(xb, vfr_b, vfi_b, xr_ws, xi_ws);
  // Stage 2: 512 waves
  k_modemix<<<64, 256, 0, stream>>>(xr_ws, xi_ws, wr, wi, or_ws, oi_ws);
  // Stage 3: 4096 waves = 512 blocks x 256 threads
  k_inv_dft<<<512, 256, 0, stream>>>(or_ws, oi_ws, vir_b, vii_b, out);
}
